// TitanMini_76115410420287
// MI455X (gfx1250) — matrix operand order repaired and checked by reading, not compiled, not run on hardware
//
#include <hip/hip_runtime.h>

// ---------------------------------------------------------------------------
// TitanMini forward for MI455X (gfx1250): bf16 WMMA GEMMs, fused attention.
// GEMM staging uses async global->LDS DMA (ASYNCcnt) when the toolchain
// exposes the gfx1250 builtins; otherwise falls back to VGPR staging.
// ---------------------------------------------------------------------------

typedef __attribute__((ext_vector_type(16))) __bf16 v16bf;
typedef __attribute__((ext_vector_type(8)))  float  v8f;

#if __has_builtin(__builtin_amdgcn_global_load_async_to_lds_b128) && \
    __has_builtin(__builtin_amdgcn_s_wait_asynccnt)
#define HAVE_ASYNC_LDS 1
typedef int v4i_ __attribute__((vector_size(16)));
typedef __attribute__((address_space(1))) v4i_ gv4i;
typedef __attribute__((address_space(3))) v4i_ lv4i;
#endif

union AB16 {
    v16bf v;
    uint4 q[2];
    unsigned short s[16];
};

__device__ __forceinline__ float bf2f(unsigned short s) {
    return __uint_as_float(((unsigned)s) << 16);
}
__device__ __forceinline__ unsigned short f2bf(float f) {
    unsigned u = __float_as_uint(f);
    u += 0x7FFFu + ((u >> 16) & 1u);          // round-to-nearest-even
    return (unsigned short)(u >> 16);
}
__device__ __forceinline__ float gelu_exact(float x) {
    return 0.5f * x * (1.0f + erff(x * 0.70710678118654752f));
}
__device__ __forceinline__ v8f wmma_bf16(const AB16& a, const AB16& b, v8f c) {
    return __builtin_amdgcn_wmma_f32_16x16x32_bf16(false, a.v, false, b.v,
                                                   (short)0, c, false, false);
}

// ---------------------------------------------------------------------------
// Weight convert + transpose: src fp32 [L][K][N] -> dst bf16 [L][N][K]
// ---------------------------------------------------------------------------
__global__ void convt_kernel(const float* __restrict__ src,
                             unsigned short* __restrict__ dst,
                             int K, int N, long total) {
    long idx = (long)blockIdx.x * 256 + threadIdx.x;
    if (idx >= total) return;
    long kn = (long)K * N;
    int  l  = (int)(idx / kn);
    long r  = idx - (long)l * kn;
    int  k  = (int)(r / N);
    int  n  = (int)(r - (long)k * N);
    dst[(long)l * kn + (long)n * K + k] = f2bf(src[idx]);
}

// ---------------------------------------------------------------------------
// Embedding + positional tables: t[b][tok][d], tok 0 = CLS
// ---------------------------------------------------------------------------
__global__ __launch_bounds__(256) void embed_kernel(
    const float* __restrict__ x, const float* __restrict__ embW,
    const float* __restrict__ embB, const float* __restrict__ cls,
    const float* __restrict__ abspos, const float* __restrict__ ftab,
    const float* __restrict__ rtab, const float* __restrict__ dtab,
    const float* __restrict__ atab, float* __restrict__ t) {
    const int tok = blockIdx.x % 65;
    const int b   = blockIdx.x / 65;
    const int tid = threadIdx.x;
#pragma unroll
    for (int i = 0; i < 2; ++i) {
        int d = tid + i * 256;
        float val;
        if (tok == 0) {
            val = cls[d];
        } else {
            int s = tok - 1;
            int f = s & 7, r = s >> 3;
            float e = embB[d];
#pragma unroll
            for (int p = 0; p < 16; ++p)
                e += x[((size_t)b * 64 + s) * 16 + p] * embW[p * 512 + d];
            val = e + abspos[s * 512 + d] + ftab[f * 512 + d] + rtab[r * 512 + d] +
                  dtab[(r + f) * 512 + d] + atab[(r - f + 7) * 512 + d];
        }
        t[((size_t)b * 65 + tok) * 512 + d] = val;
    }
}

// ---------------------------------------------------------------------------
// LayerNorm over D=512: fp32 in, bf16 out. One block per token.
// ---------------------------------------------------------------------------
__global__ __launch_bounds__(256) void ln_kernel(
    const float* __restrict__ t, const float* __restrict__ g,
    const float* __restrict__ bb, unsigned short* __restrict__ h) {
    __shared__ float red[256];
    const size_t row = blockIdx.x;
    const int tid = threadIdx.x;
    const float x0 = t[row * 512 + tid];
    const float x1 = t[row * 512 + tid + 256];
    red[tid] = x0 + x1;
    __syncthreads();
    for (int s = 128; s > 0; s >>= 1) {
        if (tid < s) red[tid] += red[tid + s];
        __syncthreads();
    }
    const float mean = red[0] * (1.0f / 512.0f);
    __syncthreads();
    const float d0 = x0 - mean, d1 = x1 - mean;
    red[tid] = d0 * d0 + d1 * d1;
    __syncthreads();
    for (int s = 128; s > 0; s >>= 1) {
        if (tid < s) red[tid] += red[tid + s];
        __syncthreads();
    }
    const float inv = rsqrtf(red[0] * (1.0f / 512.0f) + 1e-5f);
    h[row * 512 + tid]       = f2bf(d0 * inv * g[tid] + bb[tid]);
    h[row * 512 + tid + 256] = f2bf(d1 * inv * g[tid + 256] + bb[tid + 256]);
}

// ---------------------------------------------------------------------------
// bf16 WMMA GEMM: C[M][N] = A[M][K] @ Bt[N][K]^T + bias
// Block tile 128x64, K-step 32, 8 waves each computing a 32x32 subtile.
// Requires M%128==0, N%64==0, K%32==0.
// epi==0: write bf16 to Cb;  epi==1: Cf[m][n] += val (fp32 residual).
// Async path: double-buffered LDS, global->LDS DMA overlapped with WMMA.
// ---------------------------------------------------------------------------
#define AS_STRIDE 48
#define BS_STRIDE 48
__global__ __launch_bounds__(256) void gemm_bf16_kernel(
    const unsigned short* __restrict__ A, const unsigned short* __restrict__ Bt,
    const float* __restrict__ bias, unsigned short* __restrict__ Cb,
    float* __restrict__ Cf, int N, int K, int epi) {
    const int tid   = threadIdx.x;
    const int lane  = tid & 31;
    const int wave  = tid >> 5;
    const int wm    = wave >> 1;   // 0..3
    const int wn    = wave & 1;    // 0..1
    const int lhalf = lane & 15;
    const int lhi   = lane >> 4;
    const int bm0   = blockIdx.y * 128;
    const int bn0   = blockIdx.x * 64;

    // per-thread staging coordinates (16B chunks)
    const int arow = tid >> 2;          // A chunks: (arow, kc) and (arow+64, kc)
    const int akc  = (tid & 3) * 8;

    v8f acc[2][2];
#pragma unroll
    for (int mi = 0; mi < 2; ++mi)
#pragma unroll
        for (int ni = 0; ni < 2; ++ni)
#pragma unroll
            for (int r = 0; r < 8; ++r) acc[mi][ni][r] = 0.0f;

    auto compute_step = [&](const unsigned short* Asb, const unsigned short* Bsb) {
        AB16 af[2], bf[2];
#pragma unroll
        for (int mi = 0; mi < 2; ++mi) {
            const unsigned short* src =
                &Asb[(wm * 32 + mi * 16 + lhalf) * AS_STRIDE + lhi * 8];
            af[mi].q[0] = *(const uint4*)(src);
            af[mi].q[1] = *(const uint4*)(src + 16);
        }
#pragma unroll
        for (int ni = 0; ni < 2; ++ni) {
            const unsigned short* src =
                &Bsb[(wn * 32 + ni * 16 + lhalf) * BS_STRIDE + lhi * 8];
            bf[ni].q[0] = *(const uint4*)(src);
            bf[ni].q[1] = *(const uint4*)(src + 16);
        }
#pragma unroll
        for (int mi = 0; mi < 2; ++mi)
#pragma unroll
            for (int ni = 0; ni < 2; ++ni)
                acc[mi][ni] = wmma_bf16(af[mi], bf[ni], acc[mi][ni]);
    };

#ifdef HAVE_ASYNC_LDS
    __shared__ __align__(16) unsigned short As[2][128 * AS_STRIDE];
    __shared__ __align__(16) unsigned short Bs[2][64 * BS_STRIDE];

    auto stage = [&](unsigned short* AsD, unsigned short* BsD, int k0) {
        __builtin_amdgcn_global_load_async_to_lds_b128(
            (gv4i*)(A + (size_t)(bm0 + arow) * K + k0 + akc),
            (lv4i*)&AsD[arow * AS_STRIDE + akc], 0, 0);
        __builtin_amdgcn_global_load_async_to_lds_b128(
            (gv4i*)(A + (size_t)(bm0 + arow + 64) * K + k0 + akc),
            (lv4i*)&AsD[(arow + 64) * AS_STRIDE + akc], 0, 0);
        __builtin_amdgcn_global_load_async_to_lds_b128(
            (gv4i*)(Bt + (size_t)(bn0 + arow) * K + k0 + akc),
            (lv4i*)&BsD[arow * BS_STRIDE + akc], 0, 0);
    };

    stage(As[0], Bs[0], 0);
    for (int k0 = 0; k0 < K; k0 += 32) {
        const int cur = (k0 >> 5) & 1;
        if (k0 + 32 < K) {
            stage(As[cur ^ 1], Bs[cur ^ 1], k0 + 32);   // prefetch next tile
            __builtin_amdgcn_s_wait_asynccnt(3);        // current tile landed
        } else {
            __builtin_amdgcn_s_wait_asynccnt(0);
        }
        __syncthreads();
        compute_step(As[cur], Bs[cur]);
        __syncthreads();
    }
#else
    __shared__ __align__(16) unsigned short As[128 * AS_STRIDE];
    __shared__ __align__(16) unsigned short Bs[64 * BS_STRIDE];

    for (int k0 = 0; k0 < K; k0 += 32) {
        {
            uint4 v0 = *(const uint4*)(A + (size_t)(bm0 + arow) * K + k0 + akc);
            uint4 v1 = *(const uint4*)(A + (size_t)(bm0 + arow + 64) * K + k0 + akc);
            uint4 v2 = *(const uint4*)(Bt + (size_t)(bn0 + arow) * K + k0 + akc);
            *(uint4*)(&As[arow * AS_STRIDE + akc]) = v0;
            *(uint4*)(&As[(arow + 64) * AS_STRIDE + akc]) = v1;
            *(uint4*)(&Bs[arow * BS_STRIDE + akc]) = v2;
        }
        if (k0 + 32 < K) {
            __builtin_prefetch(A + (size_t)(bm0 + arow) * K + k0 + 32, 0, 1);
            __builtin_prefetch(Bt + (size_t)(bn0 + arow) * K + k0 + 32, 0, 1);
        }
        __syncthreads();
        compute_step(As, Bs);
        __syncthreads();
    }
#endif

#pragma unroll
    for (int mi = 0; mi < 2; ++mi)
#pragma unroll
        for (int ni = 0; ni < 2; ++ni)
#pragma unroll
            for (int r = 0; r < 8; ++r) {
                int m = bm0 + wm * 32 + mi * 16 + lhi * 8 + r;
                int n = bn0 + wn * 32 + ni * 16 + lhalf;
                float val = acc[mi][ni][r] + bias[n];
                if (epi == 0)
                    Cb[(size_t)m * N + n] = f2bf(val);
                else
                    Cf[(size_t)m * N + n] += val;
            }
}

// ---------------------------------------------------------------------------
// Fused attention per (qchunk-of-32, head, batch):
//   S = Q K^T / 8 + relbias ; softmax ; ctx = P V   (all WMMA)
// q/k/v/ctx layout: [b*65 + tok][h*64 + d] bf16
// ---------------------------------------------------------------------------
__global__ __launch_bounds__(256) void attn_kernel(
    const unsigned short* __restrict__ Q, const unsigned short* __restrict__ Kb,
    const unsigned short* __restrict__ V, const float* __restrict__ relb,
    unsigned short* __restrict__ ctx) {
    const int qc = blockIdx.x;   // 0..2 (query rows qc*32 .. qc*32+31)
    const int h  = blockIdx.y;   // 0..7
    const int b  = blockIdx.z;   // 0..1023
    const int tid   = threadIdx.x;
    const int lane  = tid & 31;
    const int wave  = tid >> 5;
    const int lhalf = lane & 15;
    const int lhi   = lane >> 4;

    __shared__ __align__(16) unsigned short Qs[32 * 72];
    __shared__ __align__(16) unsigned short Ks[80 * 72];
    __shared__ __align__(16) unsigned short Vts[64 * 104];  // [d][j] transposed
    __shared__ float          Ss[32 * 84];
    __shared__ __align__(16) unsigned short Ps[32 * 104];

    const size_t base = ((size_t)b * 65) * 512 + h * 64;

    // ---- stage Q (32x64), zero pad rows beyond token 64 ----
    {
        int row = tid >> 3, kc = (tid & 7) * 8;
        int tq  = qc * 32 + row;
        uint4 val = make_uint4(0u, 0u, 0u, 0u);
        if (tq < 65) val = *(const uint4*)(Q + base + (size_t)tq * 512 + kc);
        *(uint4*)(&Qs[row * 72 + kc]) = val;
    }
    // ---- stage K (80x64), zero pad ----
    for (int c = tid; c < 640; c += 256) {
        int j = c >> 3, kc = (c & 7) * 8;
        uint4 val = make_uint4(0u, 0u, 0u, 0u);
        if (j < 65) val = *(const uint4*)(Kb + base + (size_t)j * 512 + kc);
        *(uint4*)(&Ks[j * 72 + kc]) = val;
    }
    // ---- stage V transposed: Vts[d][j] for j<65; zero j=65..95 ----
    for (int c = tid; c < 520; c += 256) {
        int j = c >> 3, kc = (c & 7) * 8;
        uint4 val = *(const uint4*)(V + base + (size_t)j * 512 + kc);
        const unsigned short* ve = (const unsigned short*)&val;
#pragma unroll
        for (int e = 0; e < 8; ++e) Vts[(kc + e) * 104 + j] = ve[e];
    }
    for (int c = tid; c < 64 * 31; c += 256) {
        int d = c / 31, j = 65 + (c % 31);
        Vts[d * 104 + j] = 0;
    }
    __syncthreads();

    // ---- S = Q K^T : 2 M-tiles x 5 N-tiles, K-dim 64 (2 wmma steps) ----
    for (int tile = wave; tile < 10; tile += 8) {
        int mt = tile & 1;
        int nt = tile >> 1;
        v8f acc;
#pragma unroll
        for (int r = 0; r < 8; ++r) acc[r] = 0.0f;
#pragma unroll
        for (int ks = 0; ks < 64; ks += 32) {
            AB16 af, bf;
            const unsigned short* as = &Qs[(mt * 16 + lhalf) * 72 + ks + lhi * 8];
            af.q[0] = *(const uint4*)(as);
            af.q[1] = *(const uint4*)(as + 16);
            const unsigned short* bs = &Ks[(nt * 16 + lhalf) * 72 + ks + lhi * 8];
            bf.q[0] = *(const uint4*)(bs);
            bf.q[1] = *(const uint4*)(bs + 16);
            acc = wmma_bf16(af, bf, acc);
        }
#pragma unroll
        for (int r = 0; r < 8; ++r) {
            int qrow = mt * 16 + lhi * 8 + r;
            int jcol = nt * 16 + lhalf;
            int tq   = qc * 32 + qrow;
            float s  = acc[r] * 0.125f;  // 1/sqrt(64)
            if (tq >= 1 && tq < 65 && jcol >= 1 && jcol < 65) {
                int sq = tq - 1, sj = jcol - 1;
                int ri = min(7, max(-7, (sq >> 3) - (sj >> 3))) + 7;
                int fi = min(7, max(-7, (sq & 7) - (sj & 7))) + 7;
                s += relb[(ri * 15 + fi) * 8 + h];
            }
            Ss[qrow * 84 + jcol] = s;
        }
    }
    __syncthreads();

    // ---- softmax over 65 keys, write bf16 P (pad cols zeroed) ----
    if (tid < 32) {
        int qrow = tid;
        int tq   = qc * 32 + qrow;
        if (tq < 65) {
            float mx = -1e30f;
            for (int j = 0; j < 65; ++j) mx = fmaxf(mx, Ss[qrow * 84 + j]);
            float sum = 0.0f;
            for (int j = 0; j < 65; ++j) sum += __expf(Ss[qrow * 84 + j] - mx);
            float inv = 1.0f / sum;
            for (int j = 0; j < 65; ++j)
                Ps[qrow * 104 + j] = f2bf(__expf(Ss[qrow * 84 + j] - mx) * inv);
            for (int j = 65; j < 96; ++j) Ps[qrow * 104 + j] = 0;
        } else {
            for (int j = 0; j < 96; ++j) Ps[qrow * 104 + j] = 0;
        }
    }
    __syncthreads();

    // ---- ctx = P V : 2 M-tiles x 4 N-tiles, K-dim 96 (3 wmma steps) ----
    {
        int mt = wave & 1;
        int nt = wave >> 1;  // 0..3
        v8f acc;
#pragma unroll
        for (int r = 0; r < 8; ++r) acc[r] = 0.0f;
#pragma unroll
        for (int ks = 0; ks < 96; ks += 32) {
            AB16 af, bf;
            const unsigned short* as = &Ps[(mt * 16 + lhalf) * 104 + ks + lhi * 8];
            af.q[0] = *(const uint4*)(as);
            af.q[1] = *(const uint4*)(as + 16);
            const unsigned short* bs = &Vts[(nt * 16 + lhalf) * 104 + ks + lhi * 8];
            bf.q[0] = *(const uint4*)(bs);
            bf.q[1] = *(const uint4*)(bs + 16);
            acc = wmma_bf16(af, bf, acc);
        }
#pragma unroll
        for (int r = 0; r < 8; ++r) {
            int tq = qc * 32 + mt * 16 + lhi * 8 + r;
            int d  = nt * 16 + lhalf;
            if (tq < 65) ctx[base + (size_t)tq * 512 + d] = f2bf(acc[r]);
        }
    }
}

// ---------------------------------------------------------------------------
// GLU: g = gelu(p[:, :1536]) * p[:, 1536:]
// ---------------------------------------------------------------------------
__global__ void glu_kernel(const unsigned short* __restrict__ p,
                           unsigned short* __restrict__ g, long total) {
    long idx = (long)blockIdx.x * 256 + threadIdx.x;
    if (idx >= total) return;
    long row = idx / 1536;
    int  c   = (int)(idx - row * 1536);
    float x1 = bf2f(p[row * 3072 + c]);
    float x2 = bf2f(p[row * 3072 + 1536 + c]);
    g[idx] = f2bf(gelu_exact(x1) * x2);
}

// ---------------------------------------------------------------------------
// Policy head: out[b, s*72+n] = h[b,1+s,:] . polW[:,n] + polb[n]
// ---------------------------------------------------------------------------
__global__ void policy_kernel(const unsigned short* __restrict__ h,
                              const float* __restrict__ W,
                              const float* __restrict__ bias,
                              float* __restrict__ out) {
    long idx = (long)blockIdx.x * 256 + threadIdx.x;
    if (idx >= 1024L * 4608) return;
    int b = (int)(idx / 4608);
    int r = (int)(idx - (long)b * 4608);
    int s = r / 72;
    int n = r - s * 72;
    const unsigned short* hr = h + ((size_t)b * 65 + 1 + s) * 512;
    float acc = bias[n];
    for (int k = 0; k < 512; ++k) acc += bf2f(hr[k]) * W[k * 72 + n];
    out[idx] = acc;
}

// ---------------------------------------------------------------------------
// Value head: tanh(gelu(gelu(h0 @ v1) @ v2) @ v3). One block per batch row.
// ---------------------------------------------------------------------------
__global__ __launch_bounds__(256) void value_kernel(
    const unsigned short* __restrict__ h, const float* __restrict__ W1v,
    const float* __restrict__ b1v, const float* __restrict__ W2v,
    const float* __restrict__ b2v, const float* __restrict__ W3v,
    const float* __restrict__ b3v, float* __restrict__ out) {
    __shared__ float hrow[512];
    __shared__ float a1[256];
    __shared__ float a2[128];
    __shared__ float red[128];
    const int b = blockIdx.x;
    const int tid = threadIdx.x;
    hrow[tid]       = bf2f(h[((size_t)b * 65) * 512 + tid]);
    hrow[tid + 256] = bf2f(h[((size_t)b * 65) * 512 + tid + 256]);
    __syncthreads();
    {
        float acc = b1v[tid];
        for (int k = 0; k < 512; ++k) acc += hrow[k] * W1v[k * 256 + tid];
        a1[tid] = gelu_exact(acc);
    }
    __syncthreads();
    if (tid < 128) {
        float acc = b2v[tid];
        for (int k = 0; k < 256; ++k) acc += a1[k] * W2v[k * 128 + tid];
        a2[tid] = gelu_exact(acc);
    }
    __syncthreads();
    if (tid < 128) red[tid] = a2[tid] * W3v[tid];
    __syncthreads();
    for (int s = 64; s > 0; s >>= 1) {
        if (tid < s) red[tid] += red[tid + s];
        __syncthreads();
    }
    if (tid == 0) out[b] = tanhf(red[0] + b3v[0]);
}

// ---------------------------------------------------------------------------
// Host-side launch sequencing
// ---------------------------------------------------------------------------
static inline void launch_gemm(const unsigned short* A, const unsigned short* Bt,
                               const float* bias, unsigned short* Cb, float* Cf,
                               int M, int N, int K, int epi, hipStream_t stream) {
    dim3 grid(N / 64, M / 128);
    gemm_bf16_kernel<<<grid, 256, 0, stream>>>(A, Bt, bias, Cb, Cf, N, K, epi);
}

extern "C" void kernel_launch(void* const* d_in, const int* in_sizes, int n_in,
                              void* d_out, int out_size, void* d_ws, size_t ws_size,
                              hipStream_t stream) {
    (void)in_sizes; (void)n_in; (void)out_size; (void)ws_size;
    const float* x      = (const float*)d_in[0];
    const float* emb_W  = (const float*)d_in[1];
    const float* emb_b  = (const float*)d_in[2];
    const float* cls    = (const float*)d_in[3];
    const float* abspos = (const float*)d_in[4];
    const float* ftab   = (const float*)d_in[5];
    const float* rtab   = (const float*)d_in[6];
    const float* dtab   = (const float*)d_in[7];
    const float* atab   = (const float*)d_in[8];
    const float* relb   = (const float*)d_in[9];
    const float* Wq = (const float*)d_in[10]; const float* bq = (const float*)d_in[11];
    const float* Wk = (const float*)d_in[12]; const float* bk = (const float*)d_in[13];
    const float* Wv = (const float*)d_in[14]; const float* bv = (const float*)d_in[15];
    const float* Wo = (const float*)d_in[16]; const float* bo = (const float*)d_in[17];
    const float* ln1g = (const float*)d_in[18]; const float* ln1b = (const float*)d_in[19];
    const float* ln2g = (const float*)d_in[20]; const float* ln2b = (const float*)d_in[21];
    const float* W1 = (const float*)d_in[22]; const float* b1 = (const float*)d_in[23];
    const float* W2 = (const float*)d_in[24]; const float* b2 = (const float*)d_in[25];
    const float* flng = (const float*)d_in[26]; const float* flnb = (const float*)d_in[27];
    const float* polW = (const float*)d_in[28]; const float* polb = (const float*)d_in[29];
    const float* v1W = (const float*)d_in[30]; const float* v1b = (const float*)d_in[31];
    const float* v2W = (const float*)d_in[32]; const float* v2b = (const float*)d_in[33];
    const float* v3W = (const float*)d_in[34]; const float* v3b = (const float*)d_in[35];

    const long MT = 1024L * 65;  // 66560 rows (multiple of 128)
    char* ws = (char*)d_ws;
    size_t off = 0;
    auto alloc = [&](size_t bytes) -> char* {
        char* p = ws + off;
        off += (bytes + 255) & ~(size_t)255;
        return p;
    };
    unsigned short* WqT = (unsigned short*)alloc(10ull * 512 * 512 * 2);
    unsigned short* WkT = (unsigned short*)alloc(10ull * 512 * 512 * 2);
    unsigned short* WvT = (unsigned short*)alloc(10ull * 512 * 512 * 2);
    unsigned short* WoT = (unsigned short*)alloc(10ull * 512 * 512 * 2);
    unsigned short* W1T = (unsigned short*)alloc(10ull * 512 * 3072 * 2);
    unsigned short* W2T = (unsigned short*)alloc(10ull * 1536 * 512 * 2);
    float*          tb  = (float*)alloc((size_t)MT * 512 * 4);
    unsigned short* hb  = (unsigned short*)alloc((size_t)MT * 512 * 2);  // ln out / ctx
    unsigned short* qb  = (unsigned short*)alloc((size_t)MT * 512 * 2);
    unsigned short* kb  = (unsigned short*)alloc((size_t)MT * 512 * 2);
    unsigned short* vb  = (unsigned short*)alloc((size_t)MT * 512 * 2);
    unsigned short* pb  = (unsigned short*)alloc(8320ull * 3072 * 2);
    unsigned short* gb  = (unsigned short*)alloc(8320ull * 1536 * 2);

    // ---- weight conversion (bf16, transposed) ----
    {
        long t4 = 10L * 512 * 512;
        dim3 g4((unsigned)((t4 + 255) / 256));
        convt_kernel<<<g4, 256, 0, stream>>>(Wq, WqT, 512, 512, t4);
        convt_kernel<<<g4, 256, 0, stream>>>(Wk, WkT, 512, 512, t4);
        convt_kernel<<<g4, 256, 0, stream>>>(Wv, WvT, 512, 512, t4);
        convt_kernel<<<g4, 256, 0, stream>>>(Wo, WoT, 512, 512, t4);
        long t1 = 10L * 512 * 3072;
        convt_kernel<<<dim3((unsigned)((t1 + 255) / 256)), 256, 0, stream>>>(
            W1, W1T, 512, 3072, t1);
        long t2 = 10L * 1536 * 512;
        convt_kernel<<<dim3((unsigned)((t2 + 255) / 256)), 256, 0, stream>>>(
            W2, W2T, 1536, 512, t2);
    }

    // ---- embedding ----
    embed_kernel<<<dim3((unsigned)MT), 256, 0, stream>>>(
        x, emb_W, emb_b, cls, abspos, ftab, rtab, dtab, atab, tb);

    // ---- transformer layers ----
    for (int l = 0; l < 10; ++l) {
        ln_kernel<<<dim3((unsigned)MT), 256, 0, stream>>>(
            tb, ln1g + l * 512, ln1b + l * 512, hb);
        launch_gemm(hb, WqT + (size_t)l * 512 * 512, bq + l * 512, qb, nullptr,
                    (int)MT, 512, 512, 0, stream);
        launch_gemm(hb, WkT + (size_t)l * 512 * 512, bk + l * 512, kb, nullptr,
                    (int)MT, 512, 512, 0, stream);
        launch_gemm(hb, WvT + (size_t)l * 512 * 512, bv + l * 512, vb, nullptr,
                    (int)MT, 512, 512, 0, stream);
        attn_kernel<<<dim3(3, 8, 1024), 256, 0, stream>>>(
            qb, kb, vb, relb + (size_t)l * 15 * 15 * 8, hb);
        launch_gemm(hb, WoT + (size_t)l * 512 * 512, bo + l * 512, nullptr, tb,
                    (int)MT, 512, 512, 1, stream);
        ln_kernel<<<dim3((unsigned)MT), 256, 0, stream>>>(
            tb, ln2g + l * 512, ln2b + l * 512, hb);
        for (int c = 0; c < 8; ++c) {
            const size_t ro = (size_t)c * 8320;
            launch_gemm(hb + ro * 512, W1T + (size_t)l * 512 * 3072, b1 + l * 3072,
                        pb, nullptr, 8320, 3072, 512, 0, stream);
            long tg = 8320L * 1536;
            glu_kernel<<<dim3((unsigned)((tg + 255) / 256)), 256, 0, stream>>>(pb, gb, tg);
            launch_gemm(gb, W2T + (size_t)l * 1536 * 512, b2 + l * 512, nullptr,
                        tb + ro * 512, 8320, 512, 1536, 1, stream);
        }
    }

    // ---- heads ----
    ln_kernel<<<dim3((unsigned)MT), 256, 0, stream>>>(tb, flng, flnb, hb);
    long tp = 1024L * 4608;
    policy_kernel<<<dim3((unsigned)((tp + 255) / 256)), 256, 0, stream>>>(
        hb, polW, polb, (float*)d_out);
    value_kernel<<<dim3(1024), 256, 0, stream>>>(
        hb, v1W, v1b, v2W, v2b, v3W, v3b, (float*)d_out + tp);
}
